// TotalAttnBlock_50251117363267
// MI455X (gfx1250) — matrix operand order repaired and checked by reading, not compiled, not run on hardware
//
#include <hip/hip_runtime.h>
#include <hip/hip_bf16.h>
#include <math.h>

typedef _Float16 h16;
typedef __attribute__((ext_vector_type(16))) _Float16 v16h;
typedef __attribute__((ext_vector_type(8)))  _Float16 v8h;
typedef __attribute__((ext_vector_type(8)))  float    v8f;

// ---- problem constants ----
#define NB   4
#define CCH  64
#define DD   9
#define HW   1600
#define DHW  14400            // 9*40*40
#define TT   57600            // NB*DHW
#define AD1  32

// ---- workspace layout (bytes, all 256B aligned) ----
#define O_XT    0u            // xT / x1T / x2T  [T][64] f16   (7372800)
#define O_QKV1  7372800u      // QKV1 [T][128] f16 ; later Q2 here, ycm here
#define O_Q2    7372800u      // Q2 [T][64]
#define O_K2    14745600u     // K2 [T][64]
#define O_VCM   22118400u     // V2 channel-major [64][T]
#define O_YCM   7372800u      // y channel-major [64][T] (reuses Q2)
#define O_W1    29491200u     // [128][64] f16
#define O_W2    29507584u     // [192][64] f16
#define O_W3    29532160u     // [64][64] f16
#define O_ST    29540352u     // st1:8f, st2:8f, st3:512f

__device__ __forceinline__ float gelu_f(float x) {
  return 0.5f * x * (1.0f + erff(x * 0.70710678118654752440f));
}

__device__ __forceinline__ v8f wmma_f16(v16h a, v16h b, v8f c) {
  return __builtin_amdgcn_wmma_f32_16x16x32_f16(false, a, false, b, (short)0, c, false, false);
}

// A-fragment: matrix row-major [rows][ld], 16 rows starting row0, K = k0..k0+31
__device__ __forceinline__ v16h ldA(const h16* p, long ld, int row0, int k0) {
  int lane = threadIdx.x & 31;
  long r  = row0 + (lane & 15);
  int  kb = k0 + ((lane & 16) ? 16 : 0);
  const h16* q = p + r * ld + kb;
  v8h lo = *(const v8h*)(q);
  v8h hi = *(const v8h*)(q + 8);
  return __builtin_shufflevector(lo, hi, 0,1,2,3,4,5,6,7,8,9,10,11,12,13,14,15);
}

// B-fragment: B^T stored row-major [cols][ld] (K contiguous per column/token)
__device__ __forceinline__ v16h ldB(const h16* p, long ld, int col0, int k0) {
  int lane = threadIdx.x & 31;
  long c  = col0 + (lane & 15);
  int  kb = k0 + ((lane & 16) ? 16 : 0);
  return *(const v16h*)(p + c * ld + kb);
}

// ---------------- weight conversion ----------------
__global__ void k_prep_w(const float* WQ1, const float* WK1, const float* WV1,
                         const float* WQ2, const float* WK2, const float* WV2,
                         const float* W3, h16* w1, h16* w2, h16* w3) {
  for (int i = threadIdx.x; i < 128 * 64; i += blockDim.x) {
    int o = i >> 6, c = i & 63;
    float v = (o < 32) ? WQ1[o * 64 + c] : (o < 64) ? WK1[(o - 32) * 64 + c]
                                                     : WV1[(o - 64) * 64 + c];
    w1[i] = (h16)v;
  }
  for (int i = threadIdx.x; i < 192 * 64; i += blockDim.x) {
    int o = i >> 6, c = i & 63;
    float v = (o < 64) ? WQ2[o * 64 + c] : (o < 128) ? WK2[(o - 64) * 64 + c]
                                                      : WV2[(o - 128) * 64 + c];
    w2[i] = (h16)v;
  }
  for (int i = threadIdx.x; i < 64 * 64; i += blockDim.x) w3[i] = (h16)W3[i];
}

// ---------------- x (NCDHW f32) -> token-major [T][64] f16 ----------------
__global__ void k_xt(const float* x, h16* xt) {
  int i = blockIdx.x * blockDim.x + threadIdx.x;        // 3,686,400 threads
  int n = i / (CCH * DHW);
  int c = (i / DHW) & 63;
  int j = i % DHW;
  long t = (long)n * DHW + j;
  xt[t * 64 + c] = (h16)x[i];
}

// ---------------- GEMM1: [128x64] x xT -> QKV1 [T][128] (+Q1e/K1e) ----------------
__global__ void k_gemm1(const h16* W, const h16* X, const float* Q1e,
                        const float* K1e, h16* out) {
  int wid  = threadIdx.x >> 5;
  int tile = blockIdx.x * 4 + wid;                      // 8 * 3600 tiles
  int ot = tile & 7, tt = tile >> 3;
  int o0 = ot * 16, t0 = tt * 16;
  v8f acc = {};
  acc = wmma_f16(ldA(W, 64, o0, 0),  ldB(X, 64, t0, 0),  acc);
  acc = wmma_f16(ldA(W, 64, o0, 32), ldB(X, 64, t0, 32), acc);
  int lane = threadIdx.x & 31;
  int t = t0 + (lane & 15);
  int hi8 = (lane & 16) ? 8 : 0;
  h16* op = out + (long)t * 128;
  if (o0 < 32) {                                        // wave-uniform branch
    int d = (t / HW) % DD;
#pragma unroll
    for (int r = 0; r < 8; ++r) {
      int o = o0 + r + hi8;
      op[o] = (h16)(acc[r] + Q1e[o * DD + d]);
    }
  } else if (o0 < 64) {                                 // wave-uniform branch
    int d = (t / HW) % DD;
#pragma unroll
    for (int r = 0; r < 8; ++r) {
      int o = o0 + r + hi8;
      op[o] = (h16)(acc[r] + K1e[(o - 32) * DD + d]);
    }
  } else {
#pragma unroll
    for (int r = 0; r < 8; ++r) op[o0 + r + hi8] = (h16)acc[r];
  }
}

// ---------------- per-sample mean/rstd of V1 section ----------------
__global__ void k_red1(const h16* qkv1, float* st1) {
  int n = blockIdx.x;
  float s = 0.f, sq = 0.f;
  for (int f = threadIdx.x; f < DHW * 64; f += blockDim.x) {
    int t = f >> 6, c = f & 63;
    float v = (float)qkv1[((long)n * DHW + t) * 128 + 64 + c];
    s += v; sq += v * v;
  }
  __shared__ float ls[256], lq[256];
  ls[threadIdx.x] = s; lq[threadIdx.x] = sq; __syncthreads();
  for (int st = 128; st > 0; st >>= 1) {
    if (threadIdx.x < st) { ls[threadIdx.x] += ls[threadIdx.x + st];
                            lq[threadIdx.x] += lq[threadIdx.x + st]; }
    __syncthreads();
  }
  if (threadIdx.x == 0) {
    float inv = 1.0f / (DHW * 64.0f);
    float mu = ls[0] * inv;
    float var = lq[0] * inv - mu * mu;
    st1[n] = mu; st1[4 + n] = rsqrtf(var + 1e-5f);
  }
}

// ---------------- stage-1 depth attention (D=9, per token) ----------------
__global__ void k_attn1(const h16* qkv1, const float* st1, const float* g1,
                        const float* b1, h16* x1t) {
  int t = blockIdx.x * blockDim.x + threadIdx.x;        // 57600
  int n = t / DHW, rem = t % DHW, p = rem % HW;
  long base = ((long)n * DHW + p) * 128;                // token (n,e,p) = base + e*HW*128
  float q[32];
  const h16* qp = qkv1 + (long)t * 128;
#pragma unroll
  for (int a = 0; a < 32; ++a) q[a] = (float)qp[a];
  float s[9]; float mx = -1e30f;
  for (int e = 0; e < 9; ++e) {
    const h16* kp = qkv1 + base + (long)e * HW * 128 + 32;
    float acc = 0.f;
#pragma unroll
    for (int a = 0; a < 32; ++a) acc += q[a] * (float)kp[a];
    s[e] = acc * 0.17677669529663687f;                  // 1/sqrt(32)
    mx = fmaxf(mx, s[e]);
  }
  float l = 0.f;
  for (int e = 0; e < 9; ++e) { s[e] = expf(s[e] - mx); l += s[e]; }
  float inv = 1.0f / l;
  float mu = st1[n], rs = st1[4 + n];
  for (int c = 0; c < 64; ++c) {
    float acc = 0.f;
    for (int e = 0; e < 9; ++e)
      acc += s[e] * (float)qkv1[base + (long)e * HW * 128 + 64 + c];
    float sc = rs * g1[c];
    float xo = acc * inv * sc + (b1[c] - mu * sc);      // softmax weights sum to 1
    x1t[(long)t * 64 + c] = (h16)gelu_f(xo);
  }
}

// ---------------- GEMM2: [192x64] x x1T -> Q2,K2 token-major, V2 channel-major ----------------
__global__ void k_gemm2(const h16* W, const h16* X, const float* Q2e,
                        const float* K2e, h16* q2, h16* k2, h16* vcm) {
  int wid  = threadIdx.x >> 5;
  int tile = blockIdx.x * 4 + wid;                      // 12 * 3600 tiles
  int ot = tile % 12, tt = tile / 12;
  int o0 = ot * 16, t0 = tt * 16;
  v8f acc = {};
  acc = wmma_f16(ldA(W, 64, o0, 0),  ldB(X, 64, t0, 0),  acc);
  acc = wmma_f16(ldA(W, 64, o0, 32), ldB(X, 64, t0, 32), acc);
  int lane = threadIdx.x & 31;
  int t = t0 + (lane & 15);
  int hi8 = (lane & 16) ? 8 : 0;
  if (o0 < 64) {                                        // wave-uniform branch
    int p = t % HW;
#pragma unroll
    for (int r = 0; r < 8; ++r) {
      int o = o0 + r + hi8;
      q2[(long)t * 64 + o] = (h16)(acc[r] + Q2e[o * HW + p]);
    }
  } else if (o0 < 128) {                                // wave-uniform branch
    int p = t % HW;
#pragma unroll
    for (int r = 0; r < 8; ++r) {
      int o = o0 + r + hi8 - 64;
      k2[(long)t * 64 + o] = (h16)(acc[r] + K2e[o * HW + p]);
    }
  } else {
#pragma unroll
    for (int r = 0; r < 8; ++r)
      vcm[(long)(o0 + r + hi8 - 128) * TT + t] = (h16)acc[r];
  }
}

// ---------------- per-sample mean/rstd of V2 (channel-major) ----------------
__global__ void k_red2(const h16* vcm, float* st2) {
  int n = blockIdx.x;
  float s = 0.f, sq = 0.f;
  for (int f = threadIdx.x; f < DHW * 64; f += blockDim.x) {
    int c = f / DHW, t = f % DHW;
    float v = (float)vcm[(long)c * TT + (long)n * DHW + t];
    s += v; sq += v * v;
  }
  __shared__ float ls[256], lq[256];
  ls[threadIdx.x] = s; lq[threadIdx.x] = sq; __syncthreads();
  for (int st = 128; st > 0; st >>= 1) {
    if (threadIdx.x < st) { ls[threadIdx.x] += ls[threadIdx.x + st];
                            lq[threadIdx.x] += lq[threadIdx.x + st]; }
    __syncthreads();
  }
  if (threadIdx.x == 0) {
    float inv = 1.0f / (DHW * 64.0f);
    float mu = ls[0] * inv;
    float var = lq[0] * inv - mu * mu;
    st2[n] = mu; st2[4 + n] = rsqrtf(var + 1e-5f);
  }
}

// ---------------- normalize V2 in place ----------------
__global__ void k_norm2(h16* vcm, const float* st2, const float* g2, const float* b2) {
  int i = blockIdx.x * blockDim.x + threadIdx.x;        // 3,686,400
  int c = i / TT;
  int n = (i % TT) / DHW;
  float v = (float)vcm[i];
  vcm[i] = (h16)((v - st2[n]) * st2[4 + n] * g2[c] + b2[c]);
}

// online-softmax step for one 16-query set over 32 keys; returns P^T B-fragment
__device__ __forceinline__ v16h softmax_step(v8f s0, v8f s1, bool lo,
                                             float& m, float& l,
                                             v8f& O0, v8f& O1, v8f& O2, v8f& O3) {
  float p0[8], p1[8], tm = -3.0e38f;
#pragma unroll
  for (int r = 0; r < 8; ++r) {
    p0[r] = s0[r] * 0.125f; p1[r] = s1[r] * 0.125f;     // 1/sqrt(64)
    tm = fmaxf(tm, fmaxf(p0[r], p1[r]));
  }
  tm = fmaxf(tm, __shfl_xor(tm, 16, 32));
  float nm = fmaxf(m, tm);
  float cf = expf(m - nm);
  float ts = 0.f;
#pragma unroll
  for (int r = 0; r < 8; ++r) {
    p0[r] = expf(p0[r] - nm); p1[r] = expf(p1[r] - nm);
    ts += p0[r] + p1[r];
  }
  ts += __shfl_xor(ts, 16, 32);
  l = l * cf + ts; m = nm;
#pragma unroll
  for (int r = 0; r < 8; ++r) { O0[r] *= cf; O1[r] *= cf; O2[r] *= cf; O3[r] *= cf; }
  v16h bp;
#pragma unroll
  for (int r = 0; r < 8; ++r) {
    float oA = __shfl_xor(p0[r], 16, 32);
    float oB = __shfl_xor(p1[r], 16, 32);
    bp[r]     = (_Float16)(lo ? p0[r] : oB);
    bp[8 + r] = (_Float16)(lo ? oA    : p1[r]);
  }
  return bp;
}

// ---------------- stage-2 flash attention: per (n,d), 1600 tokens, dim 64 ----------------
// 2 waves/block, 32 queries/wave (two 16-query sets sharing K/V fragments)
__global__ void __launch_bounds__(64)
k_attn2(const h16* q2, const h16* k2, const h16* vcm, h16* x2t) {
  int wid = threadIdx.x >> 5, lane = threadIdx.x & 31;
  int nd = blockIdx.x / 25;
  int q0 = (blockIdx.x % 25) * 64 + wid * 32;
  long B0 = (long)nd * HW;
  const h16* qbase = q2 + B0 * 64;
  const h16* kbase = k2 + B0 * 64;
  const h16* vbase = vcm + B0;                          // + c*TT
  v16h bqA0 = ldB(qbase, 64, q0,      0), bqA1 = ldB(qbase, 64, q0,      32);
  v16h bqB0 = ldB(qbase, 64, q0 + 16, 0), bqB1 = ldB(qbase, 64, q0 + 16, 32);
  v8f OA0 = {}, OA1 = {}, OA2 = {}, OA3 = {};
  v8f OB0 = {}, OB1 = {}, OB2 = {}, OB3 = {};
  float mA = -1e30f, lA = 0.f, mB = -1e30f, lB = 0.f;
  bool lo = (lane & 16) == 0;
  for (int k0 = 0; k0 < HW; k0 += 32) {
    if (k0 + 32 < HW) {                                 // prefetch next K/V tile
      __builtin_prefetch(kbase + (long)(k0 + 32 + lane) * 64, 0, 3);
      __builtin_prefetch(vbase + (long)lane * TT + (k0 + 32), 0, 3);
      __builtin_prefetch(vbase + (long)(lane + 32) * TT + (k0 + 32), 0, 3);
    }
    // shared K fragments for both query sets
    v16h ka0 = ldA(kbase, 64, k0,      0);
    v16h ka1 = ldA(kbase, 64, k0,     32);
    v16h kb0 = ldA(kbase, 64, k0 + 16, 0);
    v16h kb1 = ldA(kbase, 64, k0 + 16,32);
    // S^T = K . Q^T  (M=key, N=query)
    v8f sA0 = {}, sA1 = {}, sB0 = {}, sB1 = {};
    sA0 = wmma_f16(ka0, bqA0, sA0); sA0 = wmma_f16(ka1, bqA1, sA0);
    sA1 = wmma_f16(kb0, bqA0, sA1); sA1 = wmma_f16(kb1, bqA1, sA1);
    sB0 = wmma_f16(ka0, bqB0, sB0); sB0 = wmma_f16(ka1, bqB1, sB0);
    sB1 = wmma_f16(kb0, bqB0, sB1); sB1 = wmma_f16(kb1, bqB1, sB1);
    v16h bpA = softmax_step(sA0, sA1, lo, mA, lA, OA0, OA1, OA2, OA3);
    v16h bpB = softmax_step(sB0, sB1, lo, mB, lB, OB0, OB1, OB2, OB3);
    // O^T += V^T . P^T   (A = V channel-major, shared across query sets)
    v16h v0 = ldA(vbase, TT,  0, k0);
    v16h v1 = ldA(vbase, TT, 16, k0);
    v16h v2 = ldA(vbase, TT, 32, k0);
    v16h v3 = ldA(vbase, TT, 48, k0);
    OA0 = wmma_f16(v0, bpA, OA0); OB0 = wmma_f16(v0, bpB, OB0);
    OA1 = wmma_f16(v1, bpA, OA1); OB1 = wmma_f16(v1, bpB, OB1);
    OA2 = wmma_f16(v2, bpA, OA2); OB2 = wmma_f16(v2, bpB, OB2);
    OA3 = wmma_f16(v3, bpA, OA3); OB3 = wmma_f16(v3, bpB, OB3);
  }
  float invA = 1.0f / lA, invB = 1.0f / lB;
  int hi8 = (lane & 16) ? 8 : 0;
  long tokA = B0 + q0 + (lane & 15);
  long tokB = tokA + 16;
#pragma unroll
  for (int r = 0; r < 8; ++r) {
    x2t[tokA * 64 +  0 + r + hi8] = (h16)gelu_f(OA0[r] * invA);
    x2t[tokA * 64 + 16 + r + hi8] = (h16)gelu_f(OA1[r] * invA);
    x2t[tokA * 64 + 32 + r + hi8] = (h16)gelu_f(OA2[r] * invA);
    x2t[tokA * 64 + 48 + r + hi8] = (h16)gelu_f(OA3[r] * invA);
    x2t[tokB * 64 +  0 + r + hi8] = (h16)gelu_f(OB0[r] * invB);
    x2t[tokB * 64 + 16 + r + hi8] = (h16)gelu_f(OB1[r] * invB);
    x2t[tokB * 64 + 32 + r + hi8] = (h16)gelu_f(OB2[r] * invB);
    x2t[tokB * 64 + 48 + r + hi8] = (h16)gelu_f(OB3[r] * invB);
  }
}

// ---------------- GEMM3: [64x64] x x2T -> y channel-major ----------------
__global__ void k_gemm3(const h16* W, const h16* X, h16* ycm) {
  int wid  = threadIdx.x >> 5;
  int tile = blockIdx.x * 4 + wid;                      // 4 * 3600 tiles
  int ot = tile & 3, tt = tile >> 2;
  int o0 = ot * 16, t0 = tt * 16;
  v8f acc = {};
  acc = wmma_f16(ldA(W, 64, o0, 0),  ldB(X, 64, t0, 0),  acc);
  acc = wmma_f16(ldA(W, 64, o0, 32), ldB(X, 64, t0, 32), acc);
  int lane = threadIdx.x & 31;
  int t = t0 + (lane & 15);
  int hi8 = (lane & 16) ? 8 : 0;
#pragma unroll
  for (int r = 0; r < 8; ++r)
    ycm[(long)(o0 + r + hi8) * TT + t] = (h16)acc[r];
}

// ---------------- instance-norm stats per (n,c) ----------------
__global__ void k_red3(const h16* ycm, float* st3) {
  int b = blockIdx.x;                                   // 256 = n*64+c
  int n = b >> 6, c = b & 63;
  const h16* p = ycm + (long)c * TT + (long)n * DHW;
  float s = 0.f, sq = 0.f;
  for (int i = threadIdx.x; i < DHW; i += blockDim.x) {
    float v = (float)p[i]; s += v; sq += v * v;
  }
  __shared__ float ls[256], lq[256];
  ls[threadIdx.x] = s; lq[threadIdx.x] = sq; __syncthreads();
  for (int st = 128; st > 0; st >>= 1) {
    if (threadIdx.x < st) { ls[threadIdx.x] += ls[threadIdx.x + st];
                            lq[threadIdx.x] += lq[threadIdx.x + st]; }
    __syncthreads();
  }
  if (threadIdx.x == 0) {
    float mu = ls[0] / (float)DHW;
    float var = lq[0] / (float)DHW - mu * mu;
    st3[b] = mu; st3[256 + b] = rsqrtf(var + 1e-5f);
  }
}

// ---------------- final: instance-norm + residual + GELU -> fp32 ----------------
__global__ void k_out(const float* x, const h16* ycm, const float* st3,
                      const float* g3, const float* b3, float* out) {
  int i = blockIdx.x * blockDim.x + threadIdx.x;        // 3,686,400 (NCDHW)
  int n = i / (CCH * DHW);
  int c = (i / DHW) & 63;
  int j = i % DHW;
  int b = n * 64 + c;
  float y = (float)ycm[(long)c * TT + (long)n * DHW + j];
  float v = (y - st3[b]) * st3[256 + b] * g3[c] + b3[c];
  out[i] = gelu_f(x[i] + v);
}

extern "C" void kernel_launch(void* const* d_in, const int* in_sizes, int n_in,
                              void* d_out, int out_size, void* d_ws, size_t ws_size,
                              hipStream_t stream) {
  const float* x   = (const float*)d_in[0];
  const float* WQ1 = (const float*)d_in[1];
  const float* Q1e = (const float*)d_in[2];
  const float* WK1 = (const float*)d_in[3];
  const float* K1e = (const float*)d_in[4];
  const float* WV1 = (const float*)d_in[5];
  const float* g1  = (const float*)d_in[6];
  const float* b1  = (const float*)d_in[7];
  const float* WQ2 = (const float*)d_in[8];
  const float* Q2e = (const float*)d_in[9];
  const float* WK2 = (const float*)d_in[10];
  const float* K2e = (const float*)d_in[11];
  const float* WV2 = (const float*)d_in[12];
  const float* g2  = (const float*)d_in[13];
  const float* b2  = (const float*)d_in[14];
  const float* W3  = (const float*)d_in[15];
  const float* g3  = (const float*)d_in[16];
  const float* b3  = (const float*)d_in[17];
  float* out = (float*)d_out;

  char* w = (char*)d_ws;
  h16* xt   = (h16*)(w + O_XT);     // also x1T, x2T
  h16* qkv1 = (h16*)(w + O_QKV1);
  h16* q2   = (h16*)(w + O_Q2);
  h16* k2   = (h16*)(w + O_K2);
  h16* vcm  = (h16*)(w + O_VCM);
  h16* ycm  = (h16*)(w + O_YCM);
  h16* w1   = (h16*)(w + O_W1);
  h16* w2   = (h16*)(w + O_W2);
  h16* w3   = (h16*)(w + O_W3);
  float* st1 = (float*)(w + O_ST);
  float* st2 = st1 + 8;
  float* st3 = st2 + 8;

  k_prep_w<<<1, 256, 0, stream>>>(WQ1, WK1, WV1, WQ2, WK2, WV2, W3, w1, w2, w3);
  k_xt<<<14400, 256, 0, stream>>>(x, xt);
  k_gemm1<<<7200, 128, 0, stream>>>(w1, xt, Q1e, K1e, qkv1);
  k_red1<<<4, 256, 0, stream>>>(qkv1, st1);
  k_attn1<<<225, 256, 0, stream>>>(qkv1, st1, g1, b1, xt);          // x1T -> xt region
  k_gemm2<<<10800, 128, 0, stream>>>(w2, xt, Q2e, K2e, q2, k2, vcm);
  k_red2<<<4, 256, 0, stream>>>(vcm, st2);
  k_norm2<<<14400, 256, 0, stream>>>(vcm, st2, g2, b2);
  k_attn2<<<900, 64, 0, stream>>>(q2, k2, vcm, xt);                 // x2T -> xt region
  k_gemm3<<<3600, 128, 0, stream>>>(w3, xt, ycm);
  k_red3<<<256, 256, 0, stream>>>(ycm, st3);
  k_out<<<14400, 256, 0, stream>>>(x, ycm, st3, g3, b3, out);
}
